// TestDecompRNN_57097295233344
// MI455X (gfx1250) — matrix operand order repaired and checked by reading, not compiled, not run on hardware
//
#include <hip/hip_runtime.h>
#include <hip/hip_bf16.h>
#include <math.h>

// ---------------- problem constants ----------------
// B=32, L=1024, C=32, D=512, SEG=16, K=25, SX=64, N=B*C=1024, G=3*D=1536
#define NB 32
#define NL 1024
#define NC 32
#define NN 1024      // B*C rows
#define DD 512
#define GG 1536
#define SXX 64

typedef __attribute__((ext_vector_type(16))) _Float16 v16h;
typedef __attribute__((ext_vector_type(8)))  _Float16 v8h;
typedef __attribute__((ext_vector_type(8)))  float    v8f;

// ---------------- workspace layout (bytes) ----------------
constexpr size_t WMAT = (size_t)GG * DD;            // 786432 elements per weight matrix
constexpr size_t OFF_SEQ   = 0;                                  // seq_last: NN f32
constexpr size_t OFF_XS    = OFF_SEQ   + 4096;                   // xs: NN*NL f32
constexpr size_t OFF_XT    = OFF_XS    + (size_t)NN*NL*4;        // xt: NN*NL f32
constexpr size_t OFF_W16   = OFF_XT    + (size_t)NN*NL*4;        // 4 weight mats f16 (contiguous)
constexpr size_t OFF_PES   = OFF_W16   + 4*WMAT*2;               // pe_s f16 NN*DD
constexpr size_t OFF_PET   = OFF_PES   + (size_t)NN*DD*2;
constexpr size_t OFF_ES    = OFF_PET   + (size_t)NN*DD*2;        // E_s f16 NN*DD
constexpr size_t OFF_ET    = OFF_ES    + (size_t)NN*DD*2;
constexpr size_t OFF_HS32  = OFF_ET    + (size_t)NN*DD*2;        // hs f32 NN*DD
constexpr size_t OFF_HT32  = OFF_HS32  + (size_t)NN*DD*4;
constexpr size_t OFF_HS16A = OFF_HT32  + (size_t)NN*DD*4;        // h f16 ping-pong buffers
constexpr size_t OFF_HS16B = OFF_HS16A + (size_t)NN*DD*2;
constexpr size_t OFF_HT16A = OFF_HS16B + (size_t)NN*DD*2;
constexpr size_t OFF_HT16B = OFF_HT16A + (size_t)NN*DD*2;

// ---------------- prep: f32->f16 weights, positional embeddings ----------------
__global__ __launch_bounds__(256)
void prep_kernel(const float* __restrict__ Wih_s, const float* __restrict__ Whh_s,
                 const float* __restrict__ Wih_t, const float* __restrict__ Whh_t,
                 const float* __restrict__ pos_s, const float* __restrict__ ch_s,
                 const float* __restrict__ pos_t, const float* __restrict__ ch_t,
                 _Float16* __restrict__ W16, _Float16* __restrict__ pe_s,
                 _Float16* __restrict__ pe_t) {
  size_t i = (size_t)blockIdx.x * 256 + threadIdx.x;
  if (i < 4 * WMAT) {
    size_t m = i / WMAT;
    size_t j = i - m * WMAT;
    const float* src = (m == 0) ? Wih_s : (m == 1) ? Whh_s : (m == 2) ? Wih_t : Whh_t;
    W16[i] = (_Float16)src[j];
  } else {
    size_t j = i - 4 * WMAT;                 // 0 .. 2*NN*DD
    int s = j >= (size_t)NN * DD;
    size_t k = j - (size_t)s * NN * DD;      // n*512 + d
    int n = (int)(k >> 9);
    int d = (int)(k & 511);
    int c = n & (NC - 1);
    const float* pos = s ? pos_t : pos_s;
    const float* ch  = s ? ch_t  : ch_s;
    float v = (d < 256) ? pos[d] : ch[c * 256 + (d - 256)];
    (s ? pe_t : pe_s)[k] = (_Float16)v;
  }
}

// ---------------- zero hidden state ----------------
__global__ __launch_bounds__(256)
void zero_kernel(float* __restrict__ hs32, float* __restrict__ ht32,
                 _Float16* __restrict__ hs16, _Float16* __restrict__ ht16) {
  int i = blockIdx.x * 256 + threadIdx.x;   // < NN*DD
  hs32[i] = 0.f; ht32[i] = 0.f;
  hs16[i] = (_Float16)0.f; ht16[i] = (_Float16)0.f;
}

// ---------------- decomposition: xc = x - last, 25-tap edge-replicated mean ----------------
__global__ __launch_bounds__(256)
void decomp_kernel(const float* __restrict__ x, float* __restrict__ xs_seg,
                   float* __restrict__ xt_seg, float* __restrict__ seqlast) {
  int n = blockIdx.x;                 // row = b*C + c
  int b = n >> 5, c = n & 31;
  __shared__ float row[NL];
  const float* xp = x + (size_t)b * NL * NC + c;
  for (int l = threadIdx.x; l < NL; l += 256) row[l] = xp[(size_t)l * NC];
  __syncthreads();
  float last = row[NL - 1];
  if (threadIdx.x == 0) seqlast[n] = last;
  for (int l = threadIdx.x; l < NL; l += 256) {
    float s = 0.f;
#pragma unroll
    for (int j = -12; j <= 12; ++j) {
      int idx = l + j;
      idx = idx < 0 ? 0 : (idx > NL - 1 ? NL - 1 : idx);
      s += row[idx];
    }
    float mean = s * (1.f / 25.f) - last;   // mean of (x - last)
    float xc = row[l] - last;
    xt_seg[(size_t)n * NL + l] = mean;
    xs_seg[(size_t)n * NL + l] = xc - mean;
  }
}

// ---------------- per-step segment embedding: relu(seg @ W_emb^T + b), f16 out ----------------
__global__ __launch_bounds__(256)
void embed_kernel(const float* __restrict__ xs_seg, const float* __restrict__ xt_seg,
                  const float* __restrict__ W_emb, const float* __restrict__ b_emb,
                  _Float16* __restrict__ E_s, _Float16* __restrict__ E_t, int t) {
  int n = blockIdx.x;
  int s = blockIdx.y;
  const float* src = (s == 0 ? xs_seg : xt_seg) + (size_t)n * NL + t * 16;
  _Float16* E = (s == 0 ? E_s : E_t) + (size_t)n * DD;
  __shared__ float seg[16];
  if (threadIdx.x < 16) seg[threadIdx.x] = src[threadIdx.x];
  __syncthreads();
  for (int d = threadIdx.x; d < DD; d += 256) {
    float acc = b_emb[d];
    const float* w = W_emb + (size_t)d * 16;
#pragma unroll
    for (int k = 0; k < 16; ++k) acc += seg[k] * w[k];
    E[d] = (_Float16)(acc > 0.f ? acc : 0.f);
  }
}

// ---------------- fused GRU step: 6 WMMA chains (gi/gh x r/z/n) + nonlinearity ----------------
// grid = (DD/32 = 16, NN/64 = 16, 2 streams); block = 256 (8 waves)
// wave (mw, dw): 16x16 output tile at rows m0=by*64+mw*16, cols d0=bx*32+dw*16
// For that tile it accumulates gi_r/gi_z/gi_n (A = E,   B = Wih rows d0/d0+512/d0+1024)
//                          and gh_r/gh_z/gh_n (A = h16, B = Whh rows same)
// then h' = (1-z)*tanh(gi_n + r*gh_n) + z*h, written to h32 and ping-pong h16 out.
__global__ __launch_bounds__(256)
void gru_step_kernel(const _Float16* __restrict__ E_s, const _Float16* __restrict__ E_t,
                     const _Float16* __restrict__ hsin, const _Float16* __restrict__ htin,
                     const _Float16* __restrict__ Wih_s16, const _Float16* __restrict__ Whh_s16,
                     const _Float16* __restrict__ Wih_t16, const _Float16* __restrict__ Whh_t16,
                     const float* __restrict__ bih_s, const float* __restrict__ bhh_s,
                     const float* __restrict__ bih_t, const float* __restrict__ bhh_t,
                     float* __restrict__ hs32, float* __restrict__ ht32,
                     _Float16* __restrict__ hsout, _Float16* __restrict__ htout) {
  const int st = blockIdx.z;   // 0 = seasonal, 1 = trend
  const _Float16* E   = st ? E_t     : E_s;
  const _Float16* hin = st ? htin    : hsin;
  const _Float16* Wih = st ? Wih_t16 : Wih_s16;
  const _Float16* Whh = st ? Whh_t16 : Whh_s16;
  const float* bi = st ? bih_t : bih_s;
  const float* bh = st ? bhh_t : bhh_s;
  float* h32 = st ? ht32 : hs32;
  _Float16* hout = st ? htout : hsout;

  const int lane = threadIdx.x & 31;
  const int wv = threadIdx.x >> 5;
  const int mw = wv >> 1;                 // 0..3
  const int dw = wv & 1;                  // 0..1
  const int m0 = blockIdx.y * 64 + mw * 16;
  const int d0 = blockIdx.x * 32 + dw * 16;
  const int hp = lane >> 4;
  const int l15 = lane & 15;

  const _Float16* aE = E   + (size_t)(m0 + l15) * DD;
  const _Float16* aH = hin + (size_t)(m0 + l15) * DD;
  // B fragment row pointers (W row-major = column of W^T); 16 contiguous K per lane-half
  const _Float16* wir = Wih + (size_t)(d0 +    0 + l15) * DD + 16 * hp;
  const _Float16* wiz = Wih + (size_t)(d0 +  512 + l15) * DD + 16 * hp;
  const _Float16* win = Wih + (size_t)(d0 + 1024 + l15) * DD + 16 * hp;
  const _Float16* whr = Whh + (size_t)(d0 +    0 + l15) * DD + 16 * hp;
  const _Float16* whz = Whh + (size_t)(d0 +  512 + l15) * DD + 16 * hp;
  const _Float16* whn = Whh + (size_t)(d0 + 1024 + l15) * DD + 16 * hp;

  v8f gir = {}, giz = {}, gin = {}, ghr = {}, ghz = {}, ghn = {};
  union { v16h v; v8h h[2]; } ae, ah;

#pragma unroll
  for (int kk = 0; kk < DD; kk += 32) {
    // A 16x32 f16 fragments: 16 contiguous K per lane-half (matches B)
    ae.h[0] = *(const v8h*)(aE + kk + 16 * hp);
    ae.h[1] = *(const v8h*)(aE + kk + 16 * hp + 8);
    ah.h[0] = *(const v8h*)(aH + kk + 16 * hp);
    ah.h[1] = *(const v8h*)(aH + kk + 16 * hp + 8);
    v16h bir = *(const v16h*)(wir + kk);
    v16h biz = *(const v16h*)(wiz + kk);
    v16h bin = *(const v16h*)(win + kk);
    v16h bhr = *(const v16h*)(whr + kk);
    v16h bhz = *(const v16h*)(whz + kk);
    v16h bhn = *(const v16h*)(whn + kk);
    gir = __builtin_amdgcn_wmma_f32_16x16x32_f16(false, ae.v, false, bir, (short)0, gir, false, false);
    giz = __builtin_amdgcn_wmma_f32_16x16x32_f16(false, ae.v, false, biz, (short)0, giz, false, false);
    gin = __builtin_amdgcn_wmma_f32_16x16x32_f16(false, ae.v, false, bin, (short)0, gin, false, false);
    ghr = __builtin_amdgcn_wmma_f32_16x16x32_f16(false, ah.v, false, bhr, (short)0, ghr, false, false);
    ghz = __builtin_amdgcn_wmma_f32_16x16x32_f16(false, ah.v, false, bhz, (short)0, ghz, false, false);
    ghn = __builtin_amdgcn_wmma_f32_16x16x32_f16(false, ah.v, false, bhn, (short)0, ghn, false, false);
  }

  // epilogue: C/D layout element (M = vgpr + 8*hp, N = lane&15)
  const int col = d0 + l15;
  const int rbase = m0 + 8 * hp;
  const float bir_c = bi[col];
  const float biz_c = bi[512 + col];
  const float bin_c = bi[1024 + col];
  const float bhr_c = bh[col];
  const float bhz_c = bh[512 + col];
  const float bhn_c = bh[1024 + col];
#pragma unroll
  for (int i = 0; i < 8; ++i) {
    size_t ho = (size_t)(rbase + i) * DD + col;
    float rg = 1.f / (1.f + expf(-((gir[i] + bir_c) + (ghr[i] + bhr_c))));
    float zg = 1.f / (1.f + expf(-((giz[i] + biz_c) + (ghz[i] + bhz_c))));
    float ng = tanhf((gin[i] + bin_c) + rg * (ghn[i] + bhn_c));
    float hold = h32[ho];
    float hnew = (1.f - zg) * ng + zg * hold;
    h32[ho] = hnew;
    hout[ho] = (_Float16)hnew;
  }
}

// ---------------- head: y = hs2@Wps^T + bps + ht2@Wpt^T + bpt, scatter to (B,PRED,C) ----------------
__global__ __launch_bounds__(256)
void head_kernel(const float* __restrict__ hs, const float* __restrict__ ht,
                 const float* __restrict__ Wps, const float* __restrict__ bps,
                 const float* __restrict__ Wpt, const float* __restrict__ bpt,
                 const float* __restrict__ seqlast, float* __restrict__ out) {
  int t = blockIdx.x * 256 + threadIdx.x;   // < NN*16
  int n = t >> 4, p = t & 15;
  const float* hsr = hs + (size_t)n * DD;
  const float* htr = ht + (size_t)n * DD;
  const float* w1 = Wps + (size_t)p * DD;
  const float* w2 = Wpt + (size_t)p * DD;
  float acc = bps[p] + bpt[p];
  for (int k = 0; k < DD; ++k) acc += hsr[k] * w1[k] + htr[k] * w2[k];
  int b = n >> 5, c = n & 31;
  out[(size_t)b * (16 * NC) + p * NC + c] = acc + seqlast[n];
}

// ---------------- launcher ----------------
extern "C" void kernel_launch(void* const* d_in, const int* in_sizes, int n_in,
                              void* d_out, int out_size, void* d_ws, size_t ws_size,
                              hipStream_t stream) {
  const float* x     = (const float*)d_in[0];
  const float* W_emb = (const float*)d_in[1];
  const float* b_emb = (const float*)d_in[2];
  const float* Wih_s = (const float*)d_in[3];
  const float* Whh_s = (const float*)d_in[4];
  const float* bih_s = (const float*)d_in[5];
  const float* bhh_s = (const float*)d_in[6];
  const float* Wih_t = (const float*)d_in[7];
  const float* Whh_t = (const float*)d_in[8];
  const float* bih_t = (const float*)d_in[9];
  const float* bhh_t = (const float*)d_in[10];
  const float* pos_s = (const float*)d_in[11];
  const float* ch_s  = (const float*)d_in[12];
  const float* pos_t = (const float*)d_in[13];
  const float* ch_t  = (const float*)d_in[14];
  const float* Wps   = (const float*)d_in[15];
  const float* bps   = (const float*)d_in[16];
  const float* Wpt   = (const float*)d_in[17];
  const float* bpt   = (const float*)d_in[18];
  float* out = (float*)d_out;

  char* ws = (char*)d_ws;
  float*    seqlast = (float*)(ws + OFF_SEQ);
  float*    xs_seg  = (float*)(ws + OFF_XS);
  float*    xt_seg  = (float*)(ws + OFF_XT);
  _Float16* W16     = (_Float16*)(ws + OFF_W16);
  _Float16* Wih_s16 = W16 + 0 * WMAT;
  _Float16* Whh_s16 = W16 + 1 * WMAT;
  _Float16* Wih_t16 = W16 + 2 * WMAT;
  _Float16* Whh_t16 = W16 + 3 * WMAT;
  _Float16* pe_s16  = (_Float16*)(ws + OFF_PES);
  _Float16* pe_t16  = (_Float16*)(ws + OFF_PET);
  _Float16* E_s     = (_Float16*)(ws + OFF_ES);
  _Float16* E_t     = (_Float16*)(ws + OFF_ET);
  float*    hs32    = (float*)(ws + OFF_HS32);
  float*    ht32    = (float*)(ws + OFF_HT32);
  _Float16* hs16buf[2] = { (_Float16*)(ws + OFF_HS16A), (_Float16*)(ws + OFF_HS16B) };
  _Float16* ht16buf[2] = { (_Float16*)(ws + OFF_HT16A), (_Float16*)(ws + OFF_HT16B) };

  // prep (weights f16, positional embeddings): 4*WMAT + 2*NN*DD threads
  {
    size_t tot = 4 * WMAT + 2 * (size_t)NN * DD;
    prep_kernel<<<dim3((unsigned)((tot + 255) / 256)), 256, 0, stream>>>(
        Wih_s, Whh_s, Wih_t, Whh_t, pos_s, ch_s, pos_t, ch_t, W16, pe_s16, pe_t16);
  }
  zero_kernel<<<dim3(NN * DD / 256), 256, 0, stream>>>(hs32, ht32, hs16buf[0], ht16buf[0]);
  decomp_kernel<<<dim3(NN), 256, 0, stream>>>(x, xs_seg, xt_seg, seqlast);

  dim3 sgrid(DD / 32, NN / 64, 2);     // (16, 16, 2)
  for (int t = 0; t <= SXX; ++t) {     // 64 encoder steps + 1 decoder step
    int pin = t & 1, pout = pin ^ 1;
    const _Float16* Es;
    const _Float16* Et;
    if (t < SXX) {
      embed_kernel<<<dim3(NN, 2), 256, 0, stream>>>(xs_seg, xt_seg, W_emb, b_emb, E_s, E_t, t);
      Es = E_s; Et = E_t;
    } else {
      Es = pe_s16; Et = pe_t16;        // decoder uses positional/channel embedding
    }
    gru_step_kernel<<<sgrid, 256, 0, stream>>>(
        Es, Et, hs16buf[pin], ht16buf[pin],
        Wih_s16, Whh_s16, Wih_t16, Whh_t16,
        bih_s, bhh_s, bih_t, bhh_t,
        hs32, ht32, hs16buf[pout], ht16buf[pout]);
  }

  head_kernel<<<dim3(NN * 16 / 256), 256, 0, stream>>>(hs32, ht32, Wps, bps, Wpt, bpt,
                                                       seqlast, out);
}
